// MPNN_14628658610614
// MI455X (gfx1250) — matrix operand order repaired and checked by reading, not compiled, not run on hardware
//
#include <hip/hip_runtime.h>
#include <hip/hip_bf16.h>

#define NN 100000
#define NE 600000
#define HID 128

typedef __bf16 bf16;
typedef __attribute__((ext_vector_type(16))) __bf16 v16bf;
typedef __attribute__((ext_vector_type(8)))  __bf16 v8bf;
typedef __attribute__((ext_vector_type(8)))  float  v8f;

#define SA_STR  392   // >= 384 cols, multiple of 8 (keeps 16B alignment for b128 LDS loads)
#define SA2_STR 264   // >= 256 cols
#define SH_STR  136   // >= 128 cols

// ---- WMMA fragment helpers (CDNA5 16x16x32 bf16, wave32 layouts per ISA 7.12.2) ----

// A (16x32, MxK) from LDS: lane m = lane&15; ko = (lane>>4)*8; two contiguous 8-elem chunks (K=ko..ko+7, K=16+ko..)
__device__ __forceinline__ v16bf load_a_frag(const bf16* sA, int lda, int kb, int lane) {
    const int m  = lane & 15;
    const int ko = (lane >> 4) << 3;
    const bf16* p = sA + m * lda + kb + ko;
    v8bf lo = *(const v8bf*)p;
    v8bf hi = *(const v8bf*)(p + 16);
    return __builtin_shufflevector(lo, hi, 0,1,2,3,4,5,6,7,8,9,10,11,12,13,14,15);
}

// B (32x16, KxN) from global weights [K x 128] row-major: lane n = lane&15; ko = (lane>>4)*8; K=ko..ko+7, K=16+ko..16+ko+7
__device__ __forceinline__ v16bf load_b_frag(const bf16* __restrict__ w, int kb, int nb, int lane) {
    const int n  = nb + (lane & 15);
    const int ko = (lane >> 4) << 3;
    v16bf b;
    #pragma unroll
    for (int s = 0; s < 16; ++s)
        b[s] = w[(size_t)(kb + ko + (s & 7) + ((s >> 3) << 4)) * HID + n];
    return b;
}

__device__ __forceinline__ v8f wave_gemm(const bf16* sA, int lda,
                                         const bf16* __restrict__ w, int K,
                                         int nb, int lane) {
    v8f acc = {};
    #pragma unroll
    for (int kb = 0; kb < K; kb += 32) {
        v16bf a = load_a_frag(sA, lda, kb, lane);
        v16bf b = load_b_frag(w, kb, nb, lane);
        acc = __builtin_amdgcn_wmma_f32_16x16x32_bf16(false, a, false, b, (short)0, acc, false, false);
    }
    return acc;
}

// C/D layout: n = nb + (lane&15); VGPR i -> M = i + 8*(lane>=16)
__device__ __forceinline__ void store_hidden_relu(v8f acc, const float* __restrict__ bias,
                                                  bf16* sH, int nb, int lane) {
    const int n  = nb + (lane & 15);
    const int mb = (lane >> 4) << 3;
    const float bv = bias[n];
    #pragma unroll
    for (int i = 0; i < 8; ++i) {
        float v = fmaxf(acc[i] + bv, 0.0f);
        sH[(mb + i) * SH_STR + n] = (bf16)v;
    }
}

// ---- utility kernels ----

__global__ void cvt_f32_bf16(const float* __restrict__ s, bf16* __restrict__ d, int n) {
    int i = blockIdx.x * blockDim.x + threadIdx.x;
    if (i < n) d[i] = (bf16)s[i];
}
__global__ void zero_f32(float* __restrict__ p, int n) {
    int i = blockIdx.x * blockDim.x + threadIdx.x;
    if (i < n) p[i] = 0.0f;
}

// ---- round 1: fused edge kernel (link MLP 160->128->128, msg MLP 192->128->128, scatter) ----

__global__ __launch_bounds__(256) void edge_round1(
    const bf16* __restrict__ xb, const int* __restrict__ ei, const float* __restrict__ ea,
    const bf16* __restrict__ w11, const float* __restrict__ b11,
    const bf16* __restrict__ w12, const float* __restrict__ b12,
    const bf16* __restrict__ wa1, const float* __restrict__ ba1,
    const bf16* __restrict__ wa2, const float* __restrict__ ba2,
    bf16* __restrict__ linkb, float* __restrict__ agg, float* __restrict__ deg)
{
    __shared__ bf16 sIn [16 * SA_STR];
    __shared__ bf16 sIn2[16 * SA2_STR];
    __shared__ bf16 sH  [16 * SH_STR];
    __shared__ int  sRow[16], sCol[16];

    const int t = threadIdx.x, lane = t & 31, nb = (t >> 5) * 16;
    const int eb = blockIdx.x * 16;

    if (t < 16) { sRow[t] = ei[eb + t]; sCol[t] = ei[NE + eb + t]; }
    __syncthreads();

    // stage cat(x[row], x[col], edge_attr) -> sIn[16x160]; x[col] also -> sIn2[:,0:64]
    for (int idx = t; idx < 16 * 160; idx += 256) {
        const int m = idx / 160, c = idx % 160;
        bf16 v;
        if (c < 64)       v = xb[(size_t)sRow[m] * 64 + c];
        else if (c < 128) { v = xb[(size_t)sCol[m] * 64 + (c - 64)]; sIn2[m * SA2_STR + (c - 64)] = v; }
        else              v = (bf16)ea[(size_t)(eb + m) * 32 + (c - 128)];
        sIn[m * SA_STR + c] = v;
    }
    __syncthreads();

    v8f acc = wave_gemm(sIn, SA_STR, w11, 160, nb, lane);
    store_hidden_relu(acc, b11, sH, nb, lane);
    __syncthreads();

    acc = wave_gemm(sH, SH_STR, w12, 128, nb, lane);
    {   // link -> global (bf16) and sIn2[:,64:192]
        const int n = nb + (lane & 15), mb = (lane >> 4) << 3;
        const float bv = b12[n];
        #pragma unroll
        for (int i = 0; i < 8; ++i) {
            bf16 v = (bf16)(acc[i] + bv);
            linkb[(size_t)(eb + mb + i) * HID + n] = v;
            sIn2[(mb + i) * SA2_STR + 64 + n] = v;
        }
    }
    __syncthreads();

    acc = wave_gemm(sIn2, SA2_STR, wa1, 192, nb, lane);
    store_hidden_relu(acc, ba1, sH, nb, lane);
    __syncthreads();

    acc = wave_gemm(sH, SH_STR, wa2, 128, nb, lane);
    {   // msg scatter-add
        const int n = nb + (lane & 15), mb = (lane >> 4) << 3;
        const float bv = ba2[n];
        #pragma unroll
        for (int i = 0; i < 8; ++i)
            atomicAdd(&agg[(size_t)sRow[mb + i] * HID + n], acc[i] + bv);
    }
    if (t < 16) atomicAdd(&deg[sRow[t]], 1.0f);
}

// ---- round 1 node update: h = relu(MLP(cat(x, agg_mean))) ----

__global__ __launch_bounds__(256) void node_round1(
    const bf16* __restrict__ xb, const float* __restrict__ agg, const float* __restrict__ deg,
    const bf16* __restrict__ w1, const float* __restrict__ b1,
    const bf16* __restrict__ w2, const float* __restrict__ b2,
    bf16* __restrict__ hb)
{
    __shared__ bf16 sIn[16 * SA2_STR];
    __shared__ bf16 sH [16 * SH_STR];
    __shared__ float sInv[16];
    const int t = threadIdx.x, lane = t & 31, nb = (t >> 5) * 16;
    const int n0 = blockIdx.x * 16;
    if (t < 16) sInv[t] = 1.0f / fmaxf(deg[n0 + t], 1.0f);
    __syncthreads();
    for (int idx = t; idx < 16 * 192; idx += 256) {
        const int m = idx / 192, c = idx % 192;
        const int node = n0 + m;
        bf16 v;
        if (c < 64) v = xb[(size_t)node * 64 + c];
        else        v = (bf16)(agg[(size_t)node * HID + (c - 64)] * sInv[m]);
        sIn[m * SA2_STR + c] = v;
    }
    __syncthreads();
    v8f acc = wave_gemm(sIn, SA2_STR, w1, 192, nb, lane);
    store_hidden_relu(acc, b1, sH, nb, lane);
    __syncthreads();
    acc = wave_gemm(sH, SH_STR, w2, 128, nb, lane);
    {
        const int n = nb + (lane & 15), mb = (lane >> 4) << 3;
        const float bv = b2[n];
        #pragma unroll
        for (int i = 0; i < 8; ++i)
            hb[(size_t)(n0 + mb + i) * HID + n] = (bf16)fmaxf(acc[i] + bv, 0.0f);
    }
}

// ---- round 2 fused edge kernel (link2 MLP 384->128->128, msg2 MLP 256->128->128, scatter) ----

__global__ __launch_bounds__(256) void edge_round2(
    const bf16* __restrict__ hb, const int* __restrict__ ei,
    const bf16* __restrict__ w21, const float* __restrict__ b21,
    const bf16* __restrict__ w22, const float* __restrict__ b22,
    const bf16* __restrict__ wa1, const float* __restrict__ ba1,
    const bf16* __restrict__ wa2, const float* __restrict__ ba2,
    bf16* __restrict__ linkb, float* __restrict__ agg, float* __restrict__ deg)
{
    __shared__ bf16 sIn [16 * SA_STR];
    __shared__ bf16 sIn2[16 * SA2_STR];
    __shared__ bf16 sH  [16 * SH_STR];
    __shared__ int  sRow[16], sCol[16];
    const int t = threadIdx.x, lane = t & 31, nb = (t >> 5) * 16;
    const int eb = blockIdx.x * 16;
    if (t < 16) { sRow[t] = ei[eb + t]; sCol[t] = ei[NE + eb + t]; }
    __syncthreads();
    // stage cat(h[row], h[col], link) -> sIn[16x384]; h[col] also -> sIn2[:,0:128]
    for (int idx = t; idx < 16 * 384; idx += 256) {
        const int m = idx / 384, c = idx % 384;
        bf16 v;
        if (c < 128)      v = hb[(size_t)sRow[m] * HID + c];
        else if (c < 256) { v = hb[(size_t)sCol[m] * HID + (c - 128)]; sIn2[m * SA2_STR + (c - 128)] = v; }
        else              v = linkb[(size_t)(eb + m) * HID + (c - 256)];
        sIn[m * SA_STR + c] = v;
    }
    __syncthreads();

    v8f acc = wave_gemm(sIn, SA_STR, w21, 384, nb, lane);
    store_hidden_relu(acc, b21, sH, nb, lane);
    __syncthreads();

    acc = wave_gemm(sH, SH_STR, w22, 128, nb, lane);
    {   // link2 overwrites link in place and fills sIn2[:,128:256]
        const int n = nb + (lane & 15), mb = (lane >> 4) << 3;
        const float bv = b22[n];
        #pragma unroll
        for (int i = 0; i < 8; ++i) {
            bf16 v = (bf16)(acc[i] + bv);
            linkb[(size_t)(eb + mb + i) * HID + n] = v;
            sIn2[(mb + i) * SA2_STR + 128 + n] = v;
        }
    }
    __syncthreads();

    acc = wave_gemm(sIn2, SA2_STR, wa1, 256, nb, lane);
    store_hidden_relu(acc, ba1, sH, nb, lane);
    __syncthreads();

    acc = wave_gemm(sH, SH_STR, wa2, 128, nb, lane);
    {
        const int n = nb + (lane & 15), mb = (lane >> 4) << 3;
        const float bv = ba2[n];
        #pragma unroll
        for (int i = 0; i < 8; ++i)
            atomicAdd(&agg[(size_t)sRow[mb + i] * HID + n], acc[i] + bv);
    }
    if (t < 16) atomicAdd(&deg[sRow[t]], 1.0f);
}

// ---- round 2 node update: out = MLP(cat(h, agg_mean)) with final 128->1 layer ----

__global__ __launch_bounds__(256) void node_round2(
    const bf16* __restrict__ hb, const float* __restrict__ agg, const float* __restrict__ deg,
    const bf16* __restrict__ w1, const float* __restrict__ b1,
    const float* __restrict__ w2, const float* __restrict__ b2,
    float* __restrict__ out)
{
    __shared__ bf16 sIn[16 * SA2_STR];
    __shared__ bf16 sH [16 * SH_STR];
    __shared__ float sInv[16];
    __shared__ float sRed[256];
    const int t = threadIdx.x, lane = t & 31, nb = (t >> 5) * 16;
    const int n0 = blockIdx.x * 16;
    if (t < 16) sInv[t] = 1.0f / fmaxf(deg[n0 + t], 1.0f);
    __syncthreads();
    for (int idx = t; idx < 16 * 256; idx += 256) {
        const int m = idx >> 8, c = idx & 255;
        const int node = n0 + m;
        bf16 v;
        if (c < 128) v = hb[(size_t)node * HID + c];
        else         v = (bf16)(agg[(size_t)node * HID + (c - 128)] * sInv[m]);
        sIn[m * SA2_STR + c] = v;
    }
    __syncthreads();
    v8f acc = wave_gemm(sIn, SA2_STR, w1, 256, nb, lane);
    store_hidden_relu(acc, b1, sH, nb, lane);
    __syncthreads();
    {   // final 128 -> 1 (fp32 dot): 16 threads per node, 8 MACs each
        const int m = t >> 4, j = t & 15;
        float p = 0.0f;
        #pragma unroll
        for (int q = 0; q < 8; ++q) {
            const int k = j * 8 + q;
            p += (float)sH[m * SH_STR + k] * w2[k];
        }
        sRed[t] = p;
    }
    __syncthreads();
    if (t < 16) {
        float s = b2[0];
        #pragma unroll
        for (int q = 0; q < 16; ++q) s += sRed[t * 16 + q];
        out[n0 + t] = s;
    }
}

// ---- host-side launcher ----

extern "C" void kernel_launch(void* const* d_in, const int* in_sizes, int n_in,
                              void* d_out, int out_size, void* d_ws, size_t ws_size,
                              hipStream_t stream) {
    const float* x      = (const float*)d_in[0];
    const int*   ei     = (const int*)  d_in[1];
    const float* ea     = (const float*)d_in[2];
    const float* e1_w1  = (const float*)d_in[3];  const float* e1_b1 = (const float*)d_in[4];
    const float* e1_w2  = (const float*)d_in[5];  const float* e1_b2 = (const float*)d_in[6];
    const float* n1a_w1 = (const float*)d_in[7];  const float* n1a_b1= (const float*)d_in[8];
    const float* n1a_w2 = (const float*)d_in[9];  const float* n1a_b2= (const float*)d_in[10];
    const float* n1u_w1 = (const float*)d_in[11]; const float* n1u_b1= (const float*)d_in[12];
    const float* n1u_w2 = (const float*)d_in[13]; const float* n1u_b2= (const float*)d_in[14];
    const float* e2_w1  = (const float*)d_in[15]; const float* e2_b1 = (const float*)d_in[16];
    const float* e2_w2  = (const float*)d_in[17]; const float* e2_b2 = (const float*)d_in[18];
    const float* n2a_w1 = (const float*)d_in[19]; const float* n2a_b1= (const float*)d_in[20];
    const float* n2a_w2 = (const float*)d_in[21]; const float* n2a_b2= (const float*)d_in[22];
    const float* n2u_w1 = (const float*)d_in[23]; const float* n2u_b1= (const float*)d_in[24];
    const float* n2u_w2 = (const float*)d_in[25]; const float* n2u_b2= (const float*)d_in[26];
    float* out = (float*)d_out;

    char* ws = (char*)d_ws;
    size_t off = 0;
    auto carve = [&](size_t bytes) -> void* {
        void* p = ws + off;
        off += (bytes + 255) & ~(size_t)255;
        return p;
    };
    bf16*  xb    = (bf16*) carve((size_t)NN * 64  * sizeof(bf16));
    bf16*  hb    = (bf16*) carve((size_t)NN * HID * sizeof(bf16));
    bf16*  linkb = (bf16*) carve((size_t)NE * HID * sizeof(bf16));
    float* agg   = (float*)carve((size_t)NN * HID * sizeof(float));
    float* deg   = (float*)carve((size_t)NN * sizeof(float));
    bf16* b_e1w1  = (bf16*)carve(160 * 128 * sizeof(bf16));
    bf16* b_e1w2  = (bf16*)carve(128 * 128 * sizeof(bf16));
    bf16* b_n1aw1 = (bf16*)carve(192 * 128 * sizeof(bf16));
    bf16* b_n1aw2 = (bf16*)carve(128 * 128 * sizeof(bf16));
    bf16* b_n1uw1 = (bf16*)carve(192 * 128 * sizeof(bf16));
    bf16* b_n1uw2 = (bf16*)carve(128 * 128 * sizeof(bf16));
    bf16* b_e2w1  = (bf16*)carve(384 * 128 * sizeof(bf16));
    bf16* b_e2w2  = (bf16*)carve(128 * 128 * sizeof(bf16));
    bf16* b_n2aw1 = (bf16*)carve(256 * 128 * sizeof(bf16));
    bf16* b_n2aw2 = (bf16*)carve(128 * 128 * sizeof(bf16));
    bf16* b_n2uw1 = (bf16*)carve(256 * 128 * sizeof(bf16));

    auto cvt = [&](const float* s, bf16* d, int n) {
        cvt_f32_bf16<<<(n + 255) / 256, 256, 0, stream>>>(s, d, n);
    };
    cvt(x, xb, NN * 64);
    cvt(e1_w1, b_e1w1, 160 * 128);  cvt(e1_w2, b_e1w2, 128 * 128);
    cvt(n1a_w1, b_n1aw1, 192 * 128); cvt(n1a_w2, b_n1aw2, 128 * 128);
    cvt(n1u_w1, b_n1uw1, 192 * 128); cvt(n1u_w2, b_n1uw2, 128 * 128);
    cvt(e2_w1, b_e2w1, 384 * 128);   cvt(e2_w2, b_e2w2, 128 * 128);
    cvt(n2a_w1, b_n2aw1, 256 * 128); cvt(n2a_w2, b_n2aw2, 128 * 128);
    cvt(n2u_w1, b_n2uw1, 256 * 128);

    zero_f32<<<(NN * HID + 255) / 256, 256, 0, stream>>>(agg, NN * HID);
    zero_f32<<<(NN + 255) / 256, 256, 0, stream>>>(deg, NN);

    edge_round1<<<NE / 16, 256, 0, stream>>>(xb, ei, ea,
        b_e1w1, e1_b1, b_e1w2, e1_b2, b_n1aw1, n1a_b1, b_n1aw2, n1a_b2,
        linkb, agg, deg);
    node_round1<<<NN / 16, 256, 0, stream>>>(xb, agg, deg,
        b_n1uw1, n1u_b1, b_n1uw2, n1u_b2, hb);

    zero_f32<<<(NN * HID + 255) / 256, 256, 0, stream>>>(agg, NN * HID);
    zero_f32<<<(NN + 255) / 256, 256, 0, stream>>>(deg, NN);

    edge_round2<<<NE / 16, 256, 0, stream>>>(hb, ei,
        b_e2w1, e2_b1, b_e2w2, e2_b2, b_n2aw1, n2a_b1, b_n2aw2, n2a_b2,
        linkb, agg, deg);
    node_round2<<<NN / 16, 256, 0, stream>>>(hb, agg, deg,
        b_n2uw1, n2u_b1, n2u_w2, n2u_b2, out);

    (void)in_sizes; (void)n_in; (void)out_size; (void)ws_size;
}
